// GINWithPooling_87050397155996
// MI455X (gfx1250) — compile-verified, matrix operand order checked by reading, not run on hardware
//
#include <hip/hip_runtime.h>
#include <hip/hip_bf16.h>

#define D 128            // feature width (D_IN == D_H == 128)
#define TILE_ROWS 64     // rows of the node matrix per block
#define APITCH 66        // uints per LDS row (132 bf16) -> pad kills bank conflicts

typedef __attribute__((ext_vector_type(16))) __bf16   v16bf;
typedef __attribute__((ext_vector_type(8)))  float    v8f;
typedef __attribute__((ext_vector_type(8)))  unsigned v8u;

__device__ __forceinline__ unsigned short f2bf(float f) {
    unsigned u = __float_as_uint(f);
    unsigned r = (u >> 16) & 1u;          // round to nearest even
    u += 0x7fffu + r;
    return (unsigned short)(u >> 16);
}

// ---------------------------------------------------------------------------
// Repack a 128x128 row-major f32 weight into the per-lane WMMA B-fragment
// layout (bf16, 16x16x32 tiles): out[((nt*4+kt)*32+lane)*8 + j] holds the
// packed pair {W[k,n], W[k+1,n]} that lane `lane` needs in VGPR j.
// ---------------------------------------------------------------------------
__global__ __launch_bounds__(256) void pack_w_kernel(const float* __restrict__ W,
                                                     unsigned* __restrict__ out) {
    int t = blockIdx.x * 256 + threadIdx.x;     // 8192 packed uints per matrix
    if (t >= 8192) return;
    int j    = t & 7;
    int lane = (t >> 3) & 31;
    int kt   = (t >> 8) & 3;
    int nt   = (t >> 10) & 7;
    int n    = nt * 16 + (lane & 15);
    int half = lane >> 4;
    int koff = (j < 4) ? (half * 8 + 2 * j) : (16 + half * 8 + 2 * (j - 4));
    int k    = kt * 32 + koff;
    unsigned short lo = f2bf(W[(size_t)k * D + n]);
    unsigned short hi = f2bf(W[(size_t)(k + 1) * D + n]);
    out[t] = (unsigned)lo | ((unsigned)hi << 16);
}

// ---------------------------------------------------------------------------
__global__ __launch_bounds__(256) void zero4_kernel(float4* __restrict__ p, int n4) {
    int t = blockIdx.x * 256 + threadIdx.x;
    if (t < n4) p[t] = make_float4(0.f, 0.f, 0.f, 0.f);
}

// ---------------------------------------------------------------------------
// agg[dst] += h[src]; one thread per (edge, 4-feature chunk). Gathers hit L2
// (node table is 51MB < 192MB L2); scatter adds are L2-resident f32 atomics.
// ---------------------------------------------------------------------------
__global__ __launch_bounds__(256) void scatter_edges_kernel(
        const float* __restrict__ h, const int* __restrict__ src,
        const int* __restrict__ dst, float* __restrict__ agg, int nedges) {
    int t = blockIdx.x * 256 + threadIdx.x;
    if (t >= nedges * 32) return;
    int e  = t >> 5;
    int fo = (t & 31) * 4;
    int s = src[e];
    int d = dst[e];
    const float4 v = *(const float4*)(h + (size_t)s * D + fo);
    float* a = agg + (size_t)d * D + fo;
    atomicAdd(a + 0, v.x);
    atomicAdd(a + 1, v.y);
    atomicAdd(a + 2, v.z);
    atomicAdd(a + 3, v.w);
}

// ---------------------------------------------------------------------------
// Fused GIN MLP: hout = relu( relu((hin+agg) @ Wa + ba) @ Wb + bb )
// Block = 64 rows x 128 cols, 4 waves; each wave owns a 16-row strip and runs
// 8 (n-tiles) x 4 (k-steps) v_wmma_f32_16x16x32_bf16 per GEMM.
// ---------------------------------------------------------------------------
__global__ __launch_bounds__(128) void gin_mlp_kernel(
        const float* __restrict__ hin, const float* __restrict__ agg,
        const unsigned* __restrict__ wa, const float* __restrict__ ba,
        const unsigned* __restrict__ wb, const float* __restrict__ bb,
        float* __restrict__ hout, int nrows) {
    __shared__ unsigned As[TILE_ROWS * APITCH];   // z in bf16
    __shared__ unsigned Zs[TILE_ROWS * APITCH];   // relu(z@Wa+ba) in bf16

    const int tid  = threadIdx.x;
    const int row0 = blockIdx.x * TILE_ROWS;

    // Stage z = hin + agg into LDS as bf16 (pitch-padded).
    for (int idx = tid; idx < TILE_ROWS * (D / 2); idx += 128) {
        int r = idx >> 6;          // row in tile
        int cp = idx & 63;         // column pair
        int grow = row0 + r;
        unsigned packed = 0u;
        if (grow < nrows) {
            const float2 a = *(const float2*)(hin + (size_t)grow * D + cp * 2);
            const float2 b = *(const float2*)(agg + (size_t)grow * D + cp * 2);
            packed = (unsigned)f2bf(a.x + b.x) | ((unsigned)f2bf(a.y + b.y) << 16);
        }
        As[r * APITCH + cp] = packed;
    }
    __syncthreads();

    const int w    = tid >> 5;
    const int lane = tid & 31;
    const int m0   = w * 16;           // this wave's row strip
    const int half = lane >> 4;
    const int mrow = m0 + (lane & 15); // A-fragment row for this lane
    const int ncol = lane & 15;        // B/C column within n-tile

    // ---- GEMM1: (z) @ Wa ----
    v8u afrag[4];
    #pragma unroll
    for (int kt = 0; kt < 4; ++kt) {
        #pragma unroll
        for (int j = 0; j < 8; ++j) {
            int koff = (j < 4) ? (half * 8 + 2 * j) : (16 + half * 8 + 2 * (j - 4));
            int k = kt * 32 + koff;
            afrag[kt][j] = As[mrow * APITCH + (k >> 1)];
        }
    }
    #pragma unroll
    for (int nt = 0; nt < 8; ++nt) {
        v8f acc = {};
        #pragma unroll
        for (int kt = 0; kt < 4; ++kt) {
            const uint4* bp = (const uint4*)(wa + (size_t)(((nt * 4 + kt) * 32 + lane) << 3));
            uint4 p0 = bp[0], p1 = bp[1];
            v8u braw = {p0.x, p0.y, p0.z, p0.w, p1.x, p1.y, p1.z, p1.w};
            acc = __builtin_amdgcn_wmma_f32_16x16x32_bf16(
                false, __builtin_bit_cast(v16bf, afrag[kt]),
                false, __builtin_bit_cast(v16bf, braw),
                (short)0, acc, false, false);
        }
        int n = nt * 16 + ncol;
        float bias = ba[n];
        #pragma unroll
        for (int v = 0; v < 8; ++v) {
            int r = m0 + half * 8 + v;
            float val = acc[v] + bias;
            val = val > 0.f ? val : 0.f;
            ((unsigned short*)Zs)[r * (APITCH * 2) + n] = f2bf(val);
        }
    }
    __syncthreads();

    // ---- GEMM2: relu(...) @ Wb ----
    #pragma unroll
    for (int kt = 0; kt < 4; ++kt) {
        #pragma unroll
        for (int j = 0; j < 8; ++j) {
            int koff = (j < 4) ? (half * 8 + 2 * j) : (16 + half * 8 + 2 * (j - 4));
            int k = kt * 32 + koff;
            afrag[kt][j] = Zs[mrow * APITCH + (k >> 1)];
        }
    }
    #pragma unroll
    for (int nt = 0; nt < 8; ++nt) {
        v8f acc = {};
        #pragma unroll
        for (int kt = 0; kt < 4; ++kt) {
            const uint4* bp = (const uint4*)(wb + (size_t)(((nt * 4 + kt) * 32 + lane) << 3));
            uint4 p0 = bp[0], p1 = bp[1];
            v8u braw = {p0.x, p0.y, p0.z, p0.w, p1.x, p1.y, p1.z, p1.w};
            acc = __builtin_amdgcn_wmma_f32_16x16x32_bf16(
                false, __builtin_bit_cast(v16bf, afrag[kt]),
                false, __builtin_bit_cast(v16bf, braw),
                (short)0, acc, false, false);
        }
        int n = nt * 16 + ncol;
        float bias = bb[n];
        #pragma unroll
        for (int v = 0; v < 8; ++v) {
            int r = row0 + m0 + half * 8 + v;
            float val = acc[v] + bias;
            val = val > 0.f ? val : 0.f;       // outer relu in reference
            if (r < nrows) hout[(size_t)r * D + n] = val;
        }
    }
}

// ---------------------------------------------------------------------------
// g[batch[i]] += h[i]
// ---------------------------------------------------------------------------
__global__ __launch_bounds__(256) void pool_kernel(const float* __restrict__ h,
                                                   const int* __restrict__ batch,
                                                   float* __restrict__ g, int nrows) {
    int t = blockIdx.x * 256 + threadIdx.x;
    if (t >= nrows * 32) return;
    int n  = t >> 5;
    int fo = (t & 31) * 4;
    int b = batch[n];
    const float4 v = *(const float4*)(h + (size_t)n * D + fo);
    float* gp = g + (size_t)b * D + fo;
    atomicAdd(gp + 0, v.x);
    atomicAdd(gp + 1, v.y);
    atomicAdd(gp + 2, v.z);
    atomicAdd(gp + 3, v.w);
}

// ---------------------------------------------------------------------------
// logits = g @ Wfc + bfc ; log_softmax. One wave32 per graph.
// ---------------------------------------------------------------------------
__global__ __launch_bounds__(32) void head_kernel(const float* __restrict__ g,
                                                  const float* __restrict__ Wfc,
                                                  const float* __restrict__ bfc,
                                                  float* __restrict__ out) {
    int gr = blockIdx.x;
    int lane = threadIdx.x;
    float gv[4];
    #pragma unroll
    for (int i = 0; i < 4; ++i) gv[i] = g[(size_t)gr * D + lane + 32 * i];

    float logits[10];
    #pragma unroll
    for (int j = 0; j < 10; ++j) {
        float p = 0.f;
        #pragma unroll
        for (int i = 0; i < 4; ++i) p += gv[i] * Wfc[(size_t)(lane + 32 * i) * 10 + j];
        #pragma unroll
        for (int off = 16; off > 0; off >>= 1) p += __shfl_down(p, off, 32);
        p = __shfl(p, 0, 32);
        logits[j] = p + bfc[j];
    }
    if (lane == 0) {
        float m = logits[0];
        #pragma unroll
        for (int j = 1; j < 10; ++j) m = fmaxf(m, logits[j]);
        float s = 0.f;
        #pragma unroll
        for (int j = 0; j < 10; ++j) s += __expf(logits[j] - m);
        float lse = m + __logf(s);
        #pragma unroll
        for (int j = 0; j < 10; ++j) out[(size_t)gr * 10 + j] = logits[j] - lse;
    }
}

// ---------------------------------------------------------------------------
extern "C" void kernel_launch(void* const* d_in, const int* in_sizes, int n_in,
                              void* d_out, int out_size, void* d_ws, size_t ws_size,
                              hipStream_t stream) {
    const float* x   = (const float*)d_in[0];
    const int*   ei  = (const int*)d_in[1];     // [2, E]
    const int*   bat = (const int*)d_in[2];
    const float* W1a = (const float*)d_in[3];
    const float* b1a = (const float*)d_in[4];
    const float* W1b = (const float*)d_in[5];
    const float* b1b = (const float*)d_in[6];
    const float* W2a = (const float*)d_in[7];
    const float* b2a = (const float*)d_in[8];
    const float* W2b = (const float*)d_in[9];
    const float* b2b = (const float*)d_in[10];
    const float* Wfc = (const float*)d_in[11];
    const float* bfc = (const float*)d_in[12];
    float* out = (float*)d_out;

    const int N = in_sizes[0] / D;
    const int E = in_sizes[1] / 2;
    const int G = out_size / 10;

    char* ws = (char*)d_ws;
    size_t nb = (size_t)N * D * sizeof(float);
    float*    agg = (float*)(ws);
    float*    h1  = (float*)(ws + nb);
    float*    h2  = (float*)(ws + 2 * nb);
    unsigned* wpk = (unsigned*)(ws + 3 * nb);            // 4 * 8192 uints
    float*    g   = (float*)(ws + 3 * nb + 4 * 8192 * sizeof(unsigned));

    // Pack weights into WMMA bf16 B-fragment layout.
    pack_w_kernel<<<32, 256, 0, stream>>>(W1a, wpk + 0 * 8192);
    pack_w_kernel<<<32, 256, 0, stream>>>(W1b, wpk + 1 * 8192);
    pack_w_kernel<<<32, 256, 0, stream>>>(W2a, wpk + 2 * 8192);
    pack_w_kernel<<<32, 256, 0, stream>>>(W2b, wpk + 3 * 8192);

    int zb   = (N * D / 4 + 255) / 256;
    int sb   = (E * 32 + 255) / 256;
    int gb   = (N + TILE_ROWS - 1) / TILE_ROWS;

    // Layer 1
    zero4_kernel<<<zb, 256, 0, stream>>>((float4*)agg, N * D / 4);
    scatter_edges_kernel<<<sb, 256, 0, stream>>>(x, ei, ei + E, agg, E);
    gin_mlp_kernel<<<gb, 128, 0, stream>>>(x, agg, wpk, b1a, wpk + 8192, b1b, h1, N);

    // Layer 2
    zero4_kernel<<<zb, 256, 0, stream>>>((float4*)agg, N * D / 4);
    scatter_edges_kernel<<<sb, 256, 0, stream>>>(h1, ei, ei + E, agg, E);
    gin_mlp_kernel<<<gb, 128, 0, stream>>>(h1, agg, wpk + 16384, b2a, wpk + 24576, b2b, h2, N);

    // Pool + classifier head
    zero4_kernel<<<(G * D / 4 + 255) / 256, 256, 0, stream>>>((float4*)g, G * D / 4);
    pool_kernel<<<(N * 32 + 255) / 256, 256, 0, stream>>>(h2, bat, g, N);
    head_kernel<<<G, 32, 0, stream>>>(g, Wfc, bfc, out);
}
